// EdgeGCNLayer_14173392077287
// MI455X (gfx1250) — compile-verified, matrix operand order checked by reading, not run on hardware
//
#include <hip/hip_runtime.h>
#include <hip/hip_bf16.h>

// Problem constants
#define BATCH 16
#define CIN   64
#define NPTS  2048
#define COUT  64
#define KNN   20
#define BN_EPS 1e-5f
#define SLOPE 0.2f
#define PTOT  (BATCH * NPTS)   // 32768 total points

typedef __attribute__((ext_vector_type(16))) __bf16 v16bf;
typedef __attribute__((ext_vector_type(8)))  __bf16 v8bf;
typedef __attribute__((ext_vector_type(8)))  float  v8f;

// ---------------------------------------------------------------------------
// CDNA5 WMMA 16x16x32 bf16 A/B fragment loader.
// ISA 16-bit A (16x32) layout: lane l (l16=l&15, half=l>>4) holds, for row M=l16:
//   halves 0..7  : K = kb + 8*half + {0..7}
//   halves 8..15 : K = kb + 16 + 8*half + {0..7}
// B (32x16) mirrors this with column N=l16. `p` = per-lane row/col base ptr.
// ---------------------------------------------------------------------------
__device__ __forceinline__ v16bf load_frag(const __bf16* __restrict__ p, int half, int kb) {
    const v8bf lo = *(const v8bf*)(p + kb + half * 8);
    const v8bf hi = *(const v8bf*)(p + kb + 16 + half * 8);
    v16bf r;
#pragma unroll
    for (int i = 0; i < 8; ++i) { r[i] = lo[i]; r[i + 8] = hi[i]; }
    return r;
}

__device__ __forceinline__ v8f wmma_bf16(v16bf a, v16bf b, v8f c) {
    return __builtin_amdgcn_wmma_f32_16x16x32_bf16(
        /*neg_a=*/false, a, /*neg_b=*/false, b,
        /*c_mod=*/(short)0, c, /*reuse_a=*/false, /*reuse_b=*/false);
}

// Branch-free sorted-ascending insertion into a 20-entry register list.
// Pure v_cndmask chains, in-place updates (no list duplication / mov storms).
#define INSERT20(dval, ival)                                                  \
    do {                                                                      \
        float _cd = (dval); int _ci = (ival);                                 \
        _Pragma("unroll")                                                     \
        for (int _j = 0; _j < KNN; ++_j) {                                    \
            const bool  _sw = _cd < dlist[_j];                                \
            const float _dj = dlist[_j];                                      \
            const int   _ij = ilist[_j];                                      \
            dlist[_j] = _sw ? _cd : _dj;                                      \
            ilist[_j] = _sw ? _ci : _ij;                                      \
            _cd = _sw ? _dj : _cd;                                            \
            _ci = _sw ? _ij : _ci;                                            \
        }                                                                     \
    } while (0)

// ---------------------------------------------------------------------------
// prep_x: x [B][C][N] f32 -> xt [B*N][C] bf16 (point-major) + sq norms
// ---------------------------------------------------------------------------
__global__ void prep_x_kernel(const float* __restrict__ x,
                              __bf16* __restrict__ xt,
                              float* __restrict__ sq) {
    const int gid = blockIdx.x * blockDim.x + threadIdx.x;   // point id 0..32767
    const int b = gid >> 11;
    const int n = gid & 2047;
    const float* xp = x + (size_t)b * CIN * NPTS + n;
    __bf16* xo = xt + (size_t)gid * CIN;
    float s = 0.0f;
#pragma unroll
    for (int c = 0; c < CIN; ++c) {
        float v = xp[(size_t)c * NPTS];
        s += v * v;
        xo[c] = (__bf16)v;
    }
    sq[gid] = s;
}

// ---------------------------------------------------------------------------
// prep_w: W [64][128] -> Amat [128][64] bf16:  rows 0..63 = W1 - W2 (x_j path: gathered neighbors),
// rows 64..127 = W2 (x_i path: center point).
// ---------------------------------------------------------------------------
__global__ void prep_w_kernel(const float* __restrict__ W,
                              __bf16* __restrict__ Amat) {
    const int gid = blockIdx.x * blockDim.x + threadIdx.x;   // 0..8191
    const int r = gid >> 6;     // 0..127
    const int c = gid & 63;
    float v;
    if (r < COUT) v = W[r * 128 + c] - W[r * 128 + 64 + c];  // A1 = W1 - W2
    else          v = W[(r - COUT) * 128 + 64 + c];          // A2 = W2
    Amat[gid] = (__bf16)v;
}

// ---------------------------------------------------------------------------
// knn: one wave per (batch, 16-row tile). Sweeps 128 column tiles with
// 2x v_wmma_f32_16x16x32_bf16 each; score = |x_c|^2 - 2<x_r,x_c> (row-constant
// term dropped; ranking-invariant). Tile staged via LDS so lane l owns
// (row = l&15, col-half = l>>4); per-lane top-20 insertion lists; halves
// merged through LDS; idx written directly into d_out's idx region.
// ---------------------------------------------------------------------------
__global__ void __launch_bounds__(32)
knn_kernel(const __bf16* __restrict__ xt,
           const float* __restrict__ sq,
           int* __restrict__ idx_out) {
    const int lane = threadIdx.x;
    const int half = lane >> 4;
    const int l16  = lane & 15;
    const int b       = blockIdx.x >> 7;    // /128
    const int rowbase = (blockIdx.x & 127) * 16;
    const int pbase   = b * NPTS;

    __shared__ float s_dist[256];
    __shared__ float s_md[16 * KNN];
    __shared__ int   s_mi[16 * KNN];

    // stationary A fragment: 16 query rows
    const __bf16* arow = xt + (size_t)(pbase + rowbase + l16) * CIN;
    const v16bf a0 = load_frag(arow, half, 0);
    const v16bf a1 = load_frag(arow, half, 32);

    float dlist[KNN];
    int   ilist[KNN];
#pragma unroll
    for (int j = 0; j < KNN; ++j) { dlist[j] = __FLT_MAX__; ilist[j] = 0; }

    for (int ct = 0; ct < NPTS / 16; ++ct) {
        const int colbase = ct * 16;
        const __bf16* brow = xt + (size_t)(pbase + colbase + l16) * CIN;
        if (ct + 1 < NPTS / 16)
            __builtin_prefetch(brow + 16 * CIN, 0, 0);   // global_prefetch_b8
        const v16bf b0 = load_frag(brow, half, 0);
        const v16bf b1 = load_frag(brow, half, 32);
        v8f acc = {};
        acc = wmma_bf16(a0, b0, acc);
        acc = wmma_bf16(a1, b1, acc);
        const float sqc = sq[pbase + colbase + l16];     // column norm (col = l16)

        __syncthreads();   // protect s_dist vs previous iteration readers
        // C/D layout: lane holds col=l16, rows M = v + 8*half
#pragma unroll
        for (int v = 0; v < 8; ++v) {
            const int m = v + half * 8;
            s_dist[m * 16 + l16] = sqc - 2.0f * acc[v];
        }
        __syncthreads();
        // reassign: this lane now scans row l16, columns half*8 .. half*8+7
#pragma unroll
        for (int j = 0; j < 8; ++j) {
            const int c = half * 8 + j;
            const float d = s_dist[l16 * 16 + c];
            if (d < dlist[KNN - 1]) {      // fast-reject: one cmp + execz skip
                INSERT20(d, colbase + c);
            }
        }
    }

    // merge the two column-halves of each row
    if (half) {
#pragma unroll
        for (int j = 0; j < KNN; ++j) {
            s_md[l16 * KNN + j] = dlist[j];
            s_mi[l16 * KNN + j] = ilist[j];
        }
    }
    __syncthreads();
    if (!half) {
#pragma unroll
        for (int j = 0; j < KNN; ++j) {
            const float d = s_md[l16 * KNN + j];
            const int  ci = s_mi[l16 * KNN + j];
            if (d < dlist[KNN - 1]) {
                INSERT20(d, ci);
            }
        }
        int* op = idx_out + (size_t)(pbase + rowbase + l16) * KNN;
#pragma unroll
        for (int j = 0; j < KNN; ++j) op[j] = ilist[j];
    }
}

// ---------------------------------------------------------------------------
// gemm: gy[col][R] for R in [0,128): R<64 -> g = inv*(A1.x), R>=64 ->
// y = inv*(A2.x)+shift  (BN folded). One wave per (16-col tile, 16-row tile).
// Column-major store (gy[col*128+R]) for coalesced gathers in reduce.
// ---------------------------------------------------------------------------
__global__ void __launch_bounds__(32)
gemm_kernel(const __bf16* __restrict__ xt,
            const __bf16* __restrict__ Amat,
            const float* __restrict__ gamma,
            const float* __restrict__ beta,
            const float* __restrict__ mean,
            const float* __restrict__ var,
            float* __restrict__ gy) {
    const int lane = threadIdx.x;
    const int half = lane >> 4;
    const int l16  = lane & 15;
    const int col   = blockIdx.x * 16 + l16;     // point 0..32767
    const int mtile = blockIdx.y;                // 0..7

    const __bf16* arow = Amat + (size_t)(mtile * 16 + l16) * CIN;
    const v16bf a0 = load_frag(arow, half, 0);
    const v16bf a1 = load_frag(arow, half, 32);
    const __bf16* brow = xt + (size_t)col * CIN;
    const v16bf b0 = load_frag(brow, half, 0);
    const v16bf b1 = load_frag(brow, half, 32);

    v8f acc = {};
    acc = wmma_bf16(a0, b0, acc);
    acc = wmma_bf16(a1, b1, acc);

    const int R0 = mtile * 16 + half * 8;        // first global row for this lane
    float res[8];
#pragma unroll
    for (int v = 0; v < 8; ++v) {
        const int R = R0 + v;
        const int o = R & (COUT - 1);
        const float inv = gamma[o] * rsqrtf(var[o] + BN_EPS);
        float val = acc[v] * inv;
        if (R >= COUT) val += beta[o] - mean[o] * inv;   // shift only on y-path
        res[v] = val;
    }
    float* dst = gy + (size_t)col * 128 + R0;
    *(float4*)(dst)     = make_float4(res[0], res[1], res[2], res[3]);
    *(float4*)(dst + 4) = make_float4(res[4], res[5], res[6], res[7]);
}

// ---------------------------------------------------------------------------
// reduce: out[b][o][n] = max_k leaky(g[o, idx[b,n,k]] + y[o,n])
// block (64 o-lanes x 4 points); gathers are o-contiguous / coalesced.
// ---------------------------------------------------------------------------
__global__ void reduce_kernel(const float* __restrict__ gy,
                              const int* __restrict__ idx_out,
                              float* __restrict__ out) {
    const int o  = threadIdx.x;                         // 0..63
    const int cn = blockIdx.x * 4 + threadIdx.y;        // 0..32767
    const int b  = cn >> 11;
    const int n  = cn & 2047;

    const float yv = gy[(size_t)cn * 128 + COUT + o];
    const int* ip = idx_out + (size_t)cn * KNN;
    float best = -__FLT_MAX__;
#pragma unroll
    for (int k = 0; k < KNN; ++k) {
        const int m = ip[k];                            // neighbor index within batch
        float v = gy[(size_t)(b * NPTS + m) * 128 + o] + yv;
        v = (v > 0.0f) ? v : SLOPE * v;
        best = fmaxf(best, v);
    }
    out[(size_t)b * COUT * NPTS + (size_t)o * NPTS + n] = best;
}

// ---------------------------------------------------------------------------
extern "C" void kernel_launch(void* const* d_in, const int* in_sizes, int n_in,
                              void* d_out, int out_size, void* d_ws, size_t ws_size,
                              hipStream_t stream) {
    const float* x     = (const float*)d_in[0];
    const float* W     = (const float*)d_in[1];
    const float* gamma = (const float*)d_in[2];
    const float* beta  = (const float*)d_in[3];
    const float* rmean = (const float*)d_in[4];
    const float* rvar  = (const float*)d_in[5];

    float* out     = (float*)d_out;                       // [B][COUT][N] f32
    int*   idx_out = (int*)(out + (size_t)BATCH * COUT * NPTS); // [B][N][K] i32

    // Workspace layout (~20.5 MB total; all 256B-aligned)
    char* ws = (char*)d_ws;
    size_t off = 0;
    __bf16* xt = (__bf16*)(ws + off);  off += (size_t)PTOT * CIN * sizeof(__bf16);
    off = (off + 255) & ~(size_t)255;
    float* sq = (float*)(ws + off);    off += (size_t)PTOT * sizeof(float);
    off = (off + 255) & ~(size_t)255;
    __bf16* Amat = (__bf16*)(ws + off); off += (size_t)128 * CIN * sizeof(__bf16);
    off = (off + 255) & ~(size_t)255;
    float* gy = (float*)(ws + off);    off += (size_t)PTOT * 128 * sizeof(float);
    (void)off; (void)ws_size;

    prep_x_kernel<<<PTOT / 256, 256, 0, stream>>>(x, xt, sq);
    prep_w_kernel<<<(128 * CIN) / 256, 256, 0, stream>>>(W, Amat);
    knn_kernel<<<dim3(BATCH * (NPTS / 16)), dim3(32), 0, stream>>>(xt, sq, idx_out);
    gemm_kernel<<<dim3(PTOT / 16, 8), dim3(32), 0, stream>>>(
        xt, Amat, gamma, beta, rmean, rvar, gy);
    reduce_kernel<<<dim3(PTOT / 4), dim3(64, 4), 0, stream>>>(gy, idx_out, out);
}
